// DPCA1D_41016937676858
// MI455X (gfx1250) — compile-verified, matrix operand order checked by reading, not run on hardware
//
#include <hip/hip_runtime.h>
#include <hip/hip_bf16.h>

// ---------------------------------------------------------------------------
// Types for CDNA5 WMMA
// ---------------------------------------------------------------------------
typedef __bf16 bf16;
typedef __attribute__((ext_vector_type(16))) __bf16 v16bf;
typedef __attribute__((ext_vector_type(8)))  __bf16 v8bf;
typedef __attribute__((ext_vector_type(8)))  float  v8f;
typedef __attribute__((ext_vector_type(4)))  int    v4i;

#define HEADS    16
#define DIM_HEAD 64
#define DIM      1024
#define INNER    1024
#define BATCH    8
#define LSEQ     4096
#define BH       (BATCH * HEADS)   // 128
#define TOPK     64

// Optional CDNA5 async global->LDS copies (ASYNCcnt path, §08_async_tensor).
#if defined(__has_builtin)
#if __has_builtin(__builtin_amdgcn_global_load_async_to_lds_b128) && \
    __has_builtin(__builtin_amdgcn_s_wait_asynccnt)
#define HAVE_ASYNC_LDS 1
#endif
#endif

#ifdef HAVE_ASYNC_LDS
typedef __attribute__((address_space(1))) v4i gv4i;   // global int4
typedef __attribute__((address_space(3))) v4i lv4i;   // LDS int4
static __device__ inline void async_copy16(const void* g, void* l) {
    __builtin_amdgcn_global_load_async_to_lds_b128((gv4i*)g, (lv4i*)l, 0, 0);
}
static __device__ inline void async_wait0() { __builtin_amdgcn_s_wait_asynccnt(0); }
#else
static __device__ inline void async_copy16(const void* g, void* l) {
    *(v8bf*)l = *(const v8bf*)g;
}
static __device__ inline void async_wait0() {}
#endif

static __device__ inline v8f wmma_bf16(v16bf a, v16bf b, v8f c) {
    // D = A(16x32) * B(32x16) + C, fp32 accumulate
    return __builtin_amdgcn_wmma_f32_16x16x32_bf16(
        /*neg_a=*/false, a, /*neg_b=*/false, b,
        /*c_mod=*/(short)0, c, /*reuse_a=*/false, /*reuse_b=*/false);
}

// Load one 16-lane-striped fragment from an LDS row stored K-innermost.
// Per §7.12.2: lane holds row (lane&15); K elements = [kh .. kh+7] and
// [kh+16 .. kh+23] where kh = k0 + (lane>>4)*8. Two 16B ds loads.
static __device__ inline v16bf ldfrag(const bf16* rowp, int kh) {
    v8bf c0 = *(const v8bf*)(rowp + kh);
    v8bf c1 = *(const v8bf*)(rowp + kh + 16);
    v16bf f;
#pragma unroll
    for (int e = 0; e < 8; ++e) { f[e] = c0[e]; f[e + 8] = c1[e]; }
    return f;
}

// ---------------------------------------------------------------------------
// fp32 -> bf16 elementwise
// ---------------------------------------------------------------------------
__global__ void cvt_bf16_kernel(const float* __restrict__ in, bf16* __restrict__ out, int n) {
    int i = blockIdx.x * 256 + threadIdx.x;
    if (i < n) out[i] = (bf16)in[i];
}

// ---------------------------------------------------------------------------
// Channel LayerNorm over C=1024 for each (b,l) column; bf16 output.
// ---------------------------------------------------------------------------
__global__ void chan_ln_kernel(const float* __restrict__ x,
                               const float* __restrict__ g,
                               const float* __restrict__ be,
                               bf16* __restrict__ out, int C, int L) {
    long col = (long)blockIdx.x * 256 + threadIdx.x;   // [0, B*L)
    long b = col / L, l = col % L;
    const float* xb = x + b * (long)C * L + l;
    float s = 0.f, s2 = 0.f;
    for (int c = 0; c < C; ++c) {
        float v = xb[(long)c * L];
        s += v; s2 += v * v;
    }
    float mean = s / C;
    float var  = fmaxf(s2 / C - mean * mean, 0.f);
    float inv  = 1.f / (sqrtf(var) + 1e-6f);
    bf16* ob = out + b * (long)C * L + l;
    for (int c = 0; c < C; ++c) {
        float v = xb[(long)c * L];
        ob[(long)c * L] = (bf16)(g[c] * (v - mean) * inv + be[c]);
    }
}

// ---------------------------------------------------------------------------
// bf16 WMMA GEMM:  C[b][o][n] = sum_k W[o][k] * X[b][k][n]   (fp32 out)
// Block = 256 threads = 8 waves -> 64(M) x 128(N) tile; K stepped by 32.
// Wave owns 4 output tiles (1 A-frag reused across 4 B-frags -> 4 WMMA/step).
// grid = (N/128, O/64, B)
// ---------------------------------------------------------------------------
__global__ void gemm_wmma_kernel(const bf16* __restrict__ W,
                                 const bf16* __restrict__ X,
                                 float* __restrict__ C,
                                 int O, int K, int N) {
    __shared__ alignas(16) bf16 sA[64][32];    // [m][k]
    __shared__ alignas(16) bf16 sB[128][32];   // [n][k]  (X staged transposed)
    const int t    = threadIdx.x;
    const int bN   = blockIdx.x * 128;
    const int bM   = blockIdx.y * 64;
    const int bb   = blockIdx.z;
    const bf16* Xb = X + (long)bb * K * N;
    float*      Cb = C + (long)bb * O * N;

    const int wave = t >> 5, lane = t & 31;
    const int wm  = (wave & 3) * 16;       // 4 M sub-tiles
    const int wnb = (wave >> 2) * 64;      // 2 N halves x 4 sub-tiles each
    const int r0 = lane & 15, half = lane >> 4;

    v8f acc[4];
#pragma unroll
    for (int j = 0; j < 4; ++j) acc[j] = (v8f){0.f,0.f,0.f,0.f,0.f,0.f,0.f,0.f};

    const int am = t >> 2, akc = (t & 3) * 8;   // A staging: 8 elems (16B) each

    for (int k0 = 0; k0 < K; k0 += 32) {
        // Stage A tile 64x32 (already K-innermost) -- async global->LDS b128
        async_copy16(&W[(long)(bM + am) * K + k0 + akc], &sA[am][akc]);
        // Stage B tile 32x128, transposing [k][n] -> [n][k]
#pragma unroll
        for (int it = 0; it < 2; ++it) {
            int idx = t + it * 256;            // [0, 512)
            int bk  = idx >> 4;                // k in [0,32)
            int bnc = (idx & 15) * 8;          // n chunk
            v8bf xv = *(const v8bf*)&Xb[(long)(k0 + bk) * N + bN + bnc];
#pragma unroll
            for (int e = 0; e < 8; ++e) sB[bnc + e][bk] = xv[e];
        }
        if (k0 + 32 < K) {
            __builtin_prefetch(&Xb[(long)(k0 + 32) * N + bN + (t & 15) * 8], 0, 3);
            __builtin_prefetch(&W[(long)(bM + am) * K + k0 + 32 + akc], 0, 3);
        }
        async_wait0();
        __syncthreads();

        const int kh = half * 8;
        v16bf a = ldfrag(&sA[wm + r0][0], kh);
#pragma unroll
        for (int j = 0; j < 4; ++j) {
            v16bf b = ldfrag(&sB[wnb + j * 16 + r0][0], kh);
            acc[j] = wmma_bf16(a, b, acc[j]);
        }
        __syncthreads();
    }

#pragma unroll
    for (int j = 0; j < 4; ++j) {
#pragma unroll
        for (int r = 0; r < 8; ++r) {
            int m = bM + wm + r + half * 8;
            int n = bN + wnb + j * 16 + r0;
            Cb[(long)m * N + n] = acc[j][r];
        }
    }
}

// ---------------------------------------------------------------------------
// In-place l2norm of 64-element head-dim columns.  grid = (L/256, BH)
// ---------------------------------------------------------------------------
__global__ void l2norm_kernel(float* __restrict__ p, int rowsPerBatch, int L) {
    int bh = blockIdx.y;
    int l  = blockIdx.x * 256 + threadIdx.x;
    int b = bh >> 4, h = bh & 15;
    float* base = p + ((long)b * rowsPerBatch + h * 64) * L + l;
    float ss = 0.f;
    for (int d = 0; d < 64; ++d) { float v = base[(long)d * L]; ss += v * v; }
    float inv = 1.f / fmaxf(sqrtf(ss), 1e-12f);
    for (int d = 0; d < 64; ++d) base[(long)d * L] *= inv;
}

// qp[bh][d] = sum_l q[bh][d][l].  grid = (64, BH)
__global__ void qprobe_kernel(const float* __restrict__ q, float* __restrict__ qp, int L) {
    __shared__ float red[256];
    int bh = blockIdx.y, d = blockIdx.x, t = threadIdx.x;
    const float* row = q + ((long)bh * 64 + d) * L;
    float s = 0.f;
    for (int i = t; i < L; i += 256) s += row[i];
    red[t] = s; __syncthreads();
    for (int st = 128; st > 0; st >>= 1) {
        if (t < st) red[t] += red[t + st];
        __syncthreads();
    }
    if (t == 0) qp[bh * 64 + d] = red[0];
}

// scores[bh][l] = sum_d qp[bh][d] * |k[bh][d][l]|.  grid = (L/256, BH)
__global__ void score_kernel(const float* __restrict__ kv, const float* __restrict__ qp,
                             float* __restrict__ scores, int L) {
    int bh = blockIdx.y;
    int l  = blockIdx.x * 256 + threadIdx.x;
    int b = bh >> 4, h = bh & 15;
    const float* kb  = kv + ((long)b * 2048 + h * 64) * L + l;
    const float* qpb = qp + bh * 64;
    float s = 0.f;
    for (int d = 0; d < 64; ++d) s += qpb[d] * fabsf(kb[(long)d * L]);
    scores[(long)bh * L + l] = s;
}

// Iterative top-64 selection per head-row.  grid = BH, 256 threads.
__global__ void topk_kernel(const float* __restrict__ scores, int* __restrict__ topidx, int L) {
    __shared__ float sS[LSEQ];
    __shared__ float rmax[256];
    __shared__ int   ridx[256];
    int bh = blockIdx.x, t = threadIdx.x;
    for (int i = t; i < L; i += 256) sS[i] = scores[(long)bh * L + i];
    __syncthreads();
    for (int it = 0; it < TOPK; ++it) {
        float m = -1e30f; int mi = 0;
        for (int i = t; i < L; i += 256) {
            float v = sS[i];
            if (v > m) { m = v; mi = i; }
        }
        rmax[t] = m; ridx[t] = mi;
        __syncthreads();
        for (int s = 128; s > 0; s >>= 1) {
            if (t < s && rmax[t + s] > rmax[t]) { rmax[t] = rmax[t + s]; ridx[t] = ridx[t + s]; }
            __syncthreads();
        }
        if (t == 0) { topidx[bh * TOPK + it] = ridx[0]; sS[ridx[0]] = -1e30f; }
        __syncthreads();
    }
}

// Gather selected keys/values to bf16: ksel[bh][j][d], vsel[bh][d][j].
__global__ void gather_kernel(const float* __restrict__ kv, const int* __restrict__ topidx,
                              bf16* __restrict__ ksel, bf16* __restrict__ vsel, int L) {
    int bh = blockIdx.x, t = threadIdx.x;
    int b = bh >> 4, h = bh & 15;
    const int*   idx = topidx + bh * TOPK;
    const float* kb  = kv + ((long)b * 2048 + h * 64) * L;
    const float* vb  = kv + ((long)b * 2048 + 1024 + h * 64) * L;
    bf16* ko = ksel + (long)bh * 4096;
    bf16* vo = vsel + (long)bh * 4096;
    for (int e = t; e < 4096; e += 256) {
        int j = e >> 6, d = e & 63;
        int l = idx[j];
        ko[j * 64 + d] = (bf16)kb[(long)d * L + l];
        vo[d * 64 + j] = (bf16)vb[(long)d * L + l];
    }
}

// ---------------------------------------------------------------------------
// Fused attention: sim = q^T ksel (WMMA) -> softmax(64 keys) -> out = attn V^T
// Output written transposed (b, INNER, L) in bf16 for the w_out GEMM.
// grid = (L/64, BH), 256 threads = 8 waves.
// ---------------------------------------------------------------------------
__global__ void attn_kernel(const float* __restrict__ q,
                            const bf16* __restrict__ ksel,
                            const bf16* __restrict__ vsel,
                            bf16* __restrict__ outp, int L) {
    __shared__ alignas(16) bf16  sQ[64][64];    // [i][d]  (reused as sOut [d][i])
    __shared__ alignas(16) bf16  sK[64][64];    // [j][d]
    __shared__ alignas(16) bf16  sV[64][64];    // [d][j]
    __shared__ alignas(16) float sSim[64][64];  // [i][j]
    __shared__ alignas(16) bf16  sP[64][64];    // softmaxed attn [i][j]

    const int bh = blockIdx.y;
    const int i0 = blockIdx.x * 64;
    const int t  = threadIdx.x;
    const float* qb = q + (long)bh * 64 * L;

    // Stage K/V tiles via async global->LDS (direct copy, fragment layouts)
    {
        const v8bf* ks = (const v8bf*)(ksel + (long)bh * 4096);
        const v8bf* vs = (const v8bf*)(vsel + (long)bh * 4096);
        v8bf* kd = (v8bf*)&sK[0][0];
        v8bf* vd = (v8bf*)&sV[0][0];
        for (int e = t; e < 512; e += 256) {
            async_copy16(ks + e, kd + e);
            async_copy16(vs + e, vd + e);
        }
    }
    // Stage q tile, fp32 -> bf16, transposed to [i][d]
    {
        int d = t >> 2, ic = (t & 3) * 16;
#pragma unroll
        for (int e = 0; e < 16; ++e)
            sQ[ic + e][d] = (bf16)qb[(long)d * L + i0 + ic + e];
    }
    async_wait0();
    __syncthreads();

    const int wave = t >> 5, lane = t & 31;
    const int r0 = lane & 15, half = lane >> 4;

    // sim tiles: 4x4 grid of 16x16, 2 per wave
#pragma unroll
    for (int s = 0; s < 2; ++s) {
        int tt = wave * 2 + s;
        int ti = (tt >> 2) * 16, tj = (tt & 3) * 16;
        v8f acc = {0.f, 0.f, 0.f, 0.f, 0.f, 0.f, 0.f, 0.f};
#pragma unroll
        for (int k0 = 0; k0 < 64; k0 += 32) {
            int kh = k0 + half * 8;
            v16bf a = ldfrag(&sQ[ti + r0][0], kh);
            v16bf b = ldfrag(&sK[tj + r0][0], kh);
            acc = wmma_bf16(a, b, acc);
        }
#pragma unroll
        for (int r = 0; r < 8; ++r)
            sSim[ti + r + half * 8][tj + r0] = acc[r];
    }
    __syncthreads();

    // softmax over 64 keys (cosine attention: no extra scale, per reference)
    if (t < 64) {
        float mx = -1e30f;
        for (int j = 0; j < 64; ++j) mx = fmaxf(mx, sSim[t][j]);
        float sum = 0.f;
        for (int j = 0; j < 64; ++j) {
            float e = __expf(sSim[t][j] - mx);
            sSim[t][j] = e; sum += e;
        }
        float inv = 1.f / sum;
        for (int j = 0; j < 64; ++j) sP[t][j] = (bf16)(sSim[t][j] * inv);
    }
    __syncthreads();

    // out tiles: M=i, N=d, K=j; write into sQ reused as [d][i]
#pragma unroll
    for (int s = 0; s < 2; ++s) {
        int tt = wave * 2 + s;
        int ti = (tt >> 2) * 16, td = (tt & 3) * 16;
        v8f acc = {0.f, 0.f, 0.f, 0.f, 0.f, 0.f, 0.f, 0.f};
#pragma unroll
        for (int k0 = 0; k0 < 64; k0 += 32) {
            int kh = k0 + half * 8;
            v16bf a = ldfrag(&sP[ti + r0][0], kh);
            v16bf b = ldfrag(&sV[td + r0][0], kh);
            acc = wmma_bf16(a, b, acc);
        }
#pragma unroll
        for (int r = 0; r < 8; ++r)
            sQ[td + r0][ti + r + half * 8] = (bf16)acc[r];
    }
    __syncthreads();

    // Coalesced global write: out[(bh*64 + d)*L + i0 + i]
    {
        bf16* ob = outp + (long)bh * 64 * L;
        int d = t >> 2, ic = (t & 3) * 16;
        v8bf* src = (v8bf*)&sQ[d][ic];
        *(v8bf*)(ob + (long)d * L + i0 + ic)     = src[0];
        *(v8bf*)(ob + (long)d * L + i0 + ic + 8) = src[1];
    }
}

// ---------------------------------------------------------------------------
// Final: out = gamma * LN(final; on_*) + LN(query_source; qn_*)
// ---------------------------------------------------------------------------
__global__ void final_kernel(const float* __restrict__ fin, const float* __restrict__ qs,
                             const float* __restrict__ og,  const float* __restrict__ obt,
                             const float* __restrict__ qg,  const float* __restrict__ qbt,
                             const float* __restrict__ gamma, float* __restrict__ out,
                             int C, int L) {
    long col = (long)blockIdx.x * 256 + threadIdx.x;
    long b = col / L, l = col % L;
    const float* fb  = fin + b * (long)C * L + l;
    const float* qsb = qs  + b * (long)C * L + l;
    float s1 = 0.f, s1q = 0.f, s2 = 0.f, s2q = 0.f;
    for (int c = 0; c < C; ++c) {
        float v = fb[(long)c * L];  s1 += v; s1q += v * v;
        float w = qsb[(long)c * L]; s2 += w; s2q += w * w;
    }
    float m1 = s1 / C, v1 = fmaxf(s1q / C - m1 * m1, 0.f), i1 = 1.f / (sqrtf(v1) + 1e-6f);
    float m2 = s2 / C, v2 = fmaxf(s2q / C - m2 * m2, 0.f), i2 = 1.f / (sqrtf(v2) + 1e-6f);
    float gs = gamma[0];
    float* o = out + b * (long)C * L + l;
    for (int c = 0; c < C; ++c) {
        float v = fb[(long)c * L], w = qsb[(long)c * L];
        float a = og[c] * (v - m1) * i1 + obt[c];
        float r = qg[c] * (w - m2) * i2 + qbt[c];
        o[(long)c * L] = gs * a + r;
    }
}

// ---------------------------------------------------------------------------
// Host launcher
// ---------------------------------------------------------------------------
extern "C" void kernel_launch(void* const* d_in, const int* in_sizes, int n_in,
                              void* d_out, int out_size, void* d_ws, size_t ws_size,
                              hipStream_t stream) {
    (void)in_sizes; (void)n_in; (void)out_size; (void)ws_size;
    const float* query_source = (const float*)d_in[0];
    const float* context      = (const float*)d_in[1];
    const float* cn_gamma     = (const float*)d_in[2];
    const float* cn_beta      = (const float*)d_in[3];
    const float* qn_gamma     = (const float*)d_in[4];
    const float* qn_beta      = (const float*)d_in[5];
    const float* on_gamma     = (const float*)d_in[6];
    const float* on_beta      = (const float*)d_in[7];
    const float* w_kv         = (const float*)d_in[8];
    const float* w_q          = (const float*)d_in[9];
    const float* w_out        = (const float*)d_in[10];
    const float* gamma        = (const float*)d_in[11];
    float* out = (float*)d_out;

    char* ws = (char*)d_ws;
    const size_t ACT  = (size_t)BATCH * DIM * LSEQ;          // 33,554,432 elems
    size_t off = 0;
    bf16*  ctx_n   = (bf16*)(ws + off); off += ACT * 2;                 // 64 MiB
    bf16*  qs_n    = (bf16*)(ws + off); off += ACT * 2;                 // 64 MiB
    bf16*  wkv_b   = (bf16*)(ws + off); off += (size_t)2048 * 1024 * 2; // 4 MiB
    bf16*  wq_b    = (bf16*)(ws + off); off += (size_t)1024 * 1024 * 2; // 2 MiB
    bf16*  wout_b  = (bf16*)(ws + off); off += (size_t)1024 * 1024 * 2; // 2 MiB
    float* kv      = (float*)(ws + off); off += (size_t)BATCH * 2048 * LSEQ * 4; // 256 MiB
    float* qbuf    = (float*)(ws + off); off += ACT * 4;                // 128 MiB
    float* qprobe  = (float*)(ws + off); off += (size_t)BH * 64 * 4;
    float* scores  = (float*)(ws + off); off += (size_t)BH * LSEQ * 4;  // 2 MiB
    int*   topidx  = (int*)  (ws + off); off += (size_t)BH * TOPK * 4;
    bf16*  ksel    = (bf16*) (ws + off); off += (size_t)BH * 64 * 64 * 2;
    bf16*  vsel    = (bf16*) (ws + off); off += (size_t)BH * 64 * 64 * 2;
    bf16*  attnout = (bf16*) (ws + off); off += ACT * 2;                // 64 MiB
    float* finbuf  = (float*)(ws + off); off += ACT * 4;                // 128 MiB

    // 1) weights -> bf16
    cvt_bf16_kernel<<<(2048 * 1024 + 255) / 256, 256, 0, stream>>>(w_kv, wkv_b, 2048 * 1024);
    cvt_bf16_kernel<<<(1024 * 1024 + 255) / 256, 256, 0, stream>>>(w_q, wq_b, 1024 * 1024);
    cvt_bf16_kernel<<<(1024 * 1024 + 255) / 256, 256, 0, stream>>>(w_out, wout_b, 1024 * 1024);

    // 2) channel layernorm -> bf16 activations
    const int nCols = BATCH * LSEQ / 256;  // 128 blocks
    chan_ln_kernel<<<nCols, 256, 0, stream>>>(context, cn_gamma, cn_beta, ctx_n, DIM, LSEQ);
    chan_ln_kernel<<<nCols, 256, 0, stream>>>(query_source, qn_gamma, qn_beta, qs_n, DIM, LSEQ);

    // 3) projections (WMMA GEMMs), 64x128 block tiles
    gemm_wmma_kernel<<<dim3(LSEQ / 128, 2048 / 64, BATCH), 256, 0, stream>>>(
        wkv_b, ctx_n, kv, 2048, DIM, LSEQ);
    gemm_wmma_kernel<<<dim3(LSEQ / 128, 1024 / 64, BATCH), 256, 0, stream>>>(
        wq_b, qs_n, qbuf, 1024, DIM, LSEQ);

    // 4) l2norm q and k (in place)
    l2norm_kernel<<<dim3(LSEQ / 256, BH), 256, 0, stream>>>(qbuf, 1024, LSEQ);
    l2norm_kernel<<<dim3(LSEQ / 256, BH), 256, 0, stream>>>(kv, 2048, LSEQ);

    // 5) probe scores + top-k + gather
    qprobe_kernel<<<dim3(64, BH), 256, 0, stream>>>(qbuf, qprobe, LSEQ);
    score_kernel<<<dim3(LSEQ / 256, BH), 256, 0, stream>>>(kv, qprobe, scores, LSEQ);
    topk_kernel<<<BH, 256, 0, stream>>>(scores, topidx, LSEQ);
    gather_kernel<<<BH, 256, 0, stream>>>(kv, topidx, ksel, vsel, LSEQ);

    // 6) fused attention (WMMA) -> (b, INNER, L) bf16
    attn_kernel<<<dim3(LSEQ / 64, BH), 256, 0, stream>>>(qbuf, ksel, vsel, attnout, LSEQ);

    // 7) output projection (WMMA GEMM)
    gemm_wmma_kernel<<<dim3(LSEQ / 128, 1024 / 64, BATCH), 256, 0, stream>>>(
        wout_b, attnout, finbuf, 1024, INNER, LSEQ);

    // 8) final layernorm + residual
    final_kernel<<<nCols, 256, 0, stream>>>(finbuf, query_source,
                                            on_gamma, on_beta, qn_gamma, qn_beta,
                                            gamma, out, DIM, LSEQ);
}
